// DeformConv_49890340110644
// MI455X (gfx1250) — matrix operand order repaired and checked by reading, not compiled, not run on hardware
//
#include <hip/hip_runtime.h>
#include <hip/hip_bf16.h>

// CDNA5 (gfx1250) implicit-GEMM 3x3 conv, bf16x3 split-precision WMMA.
// 4 output rows per block: A fragments (pre-split weights in d_ws) are loaded
// via global_load_b128 once per tap and reused across rows; B fragments are
// two ds_load_b128 each from hi/lo bf16 LDS planes laid out in WMMA lane order.

typedef __attribute__((ext_vector_type(16))) __bf16 v16bf;
typedef __attribute__((ext_vector_type(8)))  float  v8f;

#define HW       512
#define CIN      32
#define COUT     32
#define TILE_W   64
#define ROWS     4                         // output rows per block
#define HALO     (ROWS + 2)                // staged input rows

// weight image in d_ws: [2 planes][9 taps][4 ci-chunks][32 co][8 ci] bf16
#define WELEMS   (9 * 4 * 32 * 8)          // 9216 per plane
#define WCHUNK4  (WELEMS / 8)              // 1152 uint4 per plane

// input halo in LDS: [HALO rows][4 ci-chunks][66 x][8 ci] bf16 (per plane)
#define IXROW    66
#define ICHUNK   (IXROW * 8)               // 528
#define IROW     (4 * ICHUNK)              // 2112
#define IELEMS   (HALO * IROW)             // 12672 (25344 B per plane)

union Frag { uint4 q[2]; v16bf v; };

// round-to-nearest-even fp32 -> bf16 bits
__device__ __forceinline__ unsigned f2bf(float f) {
    unsigned u = __float_as_uint(f);
    return ((u + 0x7fffu + ((u >> 16) & 1u)) >> 16) & 0xffffu;
}

// ---------------- one-block prep: split weights into hi/lo planes in d_ws ----
__global__ __launch_bounds__(256) void prep_weights(
    const float* __restrict__ w,            // [32 co][32 ci][3][3]
    unsigned short* __restrict__ whi,       // [9][4][32][8]
    unsigned short* __restrict__ wlo)
{
    for (int e = threadIdx.x; e < 9 * COUT * CIN; e += 256) {
        int co = e / 288;
        int r  = e - co * 288;
        int ci = r / 9;
        int t  = r - ci * 9;
        float v = w[e];
        unsigned hi = f2bf(v);
        unsigned lo = f2bf(v - __uint_as_float(hi << 16));
        int idx = ((t * 4 + (ci >> 3)) * COUT + co) * 8 + (ci & 7);
        whi[idx] = (unsigned short)hi;
        wlo[idx] = (unsigned short)lo;
    }
}

// ---------------- main conv kernel ------------------------------------------
__global__ __launch_bounds__(256) void conv3x3_wmma_bf16x3(
    const float* __restrict__ input,        // [8][32][512][512]
    const uint4*  __restrict__ w4,          // hi plane then lo plane, uint4 view
    const float* __restrict__ bias,         // [32]
    float* __restrict__ out)                // [8][32][512][512]
{
    __shared__ __align__(16) unsigned short ihi[IELEMS];   // 25344 B
    __shared__ __align__(16) unsigned short ilo[IELEMS];   // 25344 B

    const int tid  = threadIdx.x;
    const int b    = blockIdx.x;
    const int wblk = b & 7;                 // 8 tiles of 64 px
    const int hblk = (b >> 3) & 127;        // 128 blocks of 4 rows
    const int n    = b >> 10;
    const int w0   = wblk * TILE_W;
    const int h0   = hblk * ROWS;

    // ---- stage input halo (rows h0-1..h0+4, cols w0-1..w0+64): split+transpose
    for (int e = tid; e < HALO * IXROW; e += 256) {
        const int y  = e / IXROW;
        const int x  = e - y * IXROW;
        const int gy = h0 - 1 + y;
        const int gx = w0 - 1 + x;
        const bool inr = ((unsigned)gy < (unsigned)HW) &&
                         ((unsigned)gx < (unsigned)HW);
        const float* ip = input + ((long)n * CIN * HW + gy) * HW + gx;
        const int lbase = y * IROW + x * 8;
        #pragma unroll
        for (int ci = 0; ci < CIN; ++ci) {
            float v = inr ? ip[(size_t)ci * HW * HW] : 0.0f;
            unsigned hi = f2bf(v);
            unsigned lo = f2bf(v - __uint_as_float(hi << 16));
            int li = lbase + (ci >> 3) * ICHUNK + (ci & 7);
            ihi[li] = (unsigned short)hi;
            ilo[li] = (unsigned short)lo;
        }
    }
    __syncthreads();

    const int lane  = tid & 31;
    const int wid   = tid >> 5;
    const int mtile = wid & 1;              // 16-channel tile
    const int ntile = wid >> 1;             // 16-pixel tile
    const int half  = lane >> 4;
    const int lm    = lane & 15;

    // B: lane col n=lm; frag = ci-chunks {half, 2+half} at x = x0+kw
    const int x0    = ntile * 16 + lm;
    const int bbase = x0 * 8 + half * ICHUNK;

    v8f acc[ROWS] = {};

    #pragma unroll
    for (int t = 0; t < 9; ++t) {
        const int kh = t / 3, kw = t - kh * 3;

        // A: lane row m=lm; frag = chunks {half, 2+half}; reused for all rows
        Frag ah, al;
        const int ai = (t * 4 + half) * 32 + lm;
        ah.q[0] = w4[ai];
        ah.q[1] = w4[ai + 2 * 32];
        al.q[0] = w4[WCHUNK4 + ai];
        al.q[1] = w4[WCHUNK4 + ai + 2 * 32];

        #pragma unroll
        for (int r = 0; r < ROWS; ++r) {
            Frag bh, bl;
            const int b0 = bbase + (r + kh) * IROW + kw * 8;
            bh.q[0] = *(const uint4*)(ihi + b0);
            bh.q[1] = *(const uint4*)(ihi + b0 + 2 * ICHUNK);
            bl.q[0] = *(const uint4*)(ilo + b0);
            bl.q[1] = *(const uint4*)(ilo + b0 + 2 * ICHUNK);

            // bf16x3 fp32 emulation: hi*hi + lo*hi + hi*lo, f32 accumulate
            acc[r] = __builtin_amdgcn_wmma_f32_16x16x32_bf16(
                         false, ah.v, false, bh.v, (short)0, acc[r], false, false);
            acc[r] = __builtin_amdgcn_wmma_f32_16x16x32_bf16(
                         false, al.v, false, bh.v, (short)0, acc[r], false, false);
            acc[r] = __builtin_amdgcn_wmma_f32_16x16x32_bf16(
                         false, ah.v, false, bl.v, (short)0, acc[r], false, false);
        }
    }

    // ---- D layout: lane col n=lm, row m = v + 8*half
    const int cobase = mtile * 16 + half * 8;
    const int wcol   = w0 + ntile * 16 + lm;
    float bv[8];
    #pragma unroll
    for (int v = 0; v < 8; ++v) bv[v] = bias[cobase + v];

    #pragma unroll
    for (int r = 0; r < ROWS; ++r) {
        float* op = out + (((long)n * COUT + cobase) * HW + (h0 + r)) * HW + wcol;
        #pragma unroll
        for (int v = 0; v < 8; ++v)
            op[(size_t)v * HW * HW] = acc[r][v] + bv[v];
    }
}

extern "C" void kernel_launch(void* const* d_in, const int* in_sizes, int n_in,
                              void* d_out, int out_size, void* d_ws, size_t ws_size,
                              hipStream_t stream) {
    const float* input  = (const float*)d_in[0];
    const float* weight = (const float*)d_in[1];
    const float* bias   = (const float*)d_in[2];
    float* out = (float*)d_out;

    unsigned short* whi = (unsigned short*)d_ws;
    unsigned short* wlo = whi + WELEMS;

    prep_weights<<<1, 256, 0, stream>>>(weight, whi, wlo);

    // one block per (n, 4-row band, 64-pixel tile): 8 * 128 * 8 = 8192 blocks
    dim3 grid(8 * (HW / ROWS) * (HW / TILE_W));
    conv3x3_wmma_bf16x3<<<grid, 256, 0, stream>>>(input, (const uint4*)d_ws,
                                                  bias, out);
}
